// ASTGCN_block_51625506898637
// MI455X (gfx1250) — matrix operand order repaired and checked by reading, not compiled, not run on hardware
//
#include <hip/hip_runtime.h>
#include <hip/hip_bf16.h>

// ---------------------------------------------------------------------------
// ASTGCN block for MI455X (gfx1250, wave32). Heavy GEMMs use
// v_wmma_f32_16x16x32_bf16 with 2x2 register blocking per wave
// (1 fragment load per WMMA); everything else is f32 VALU.
// ---------------------------------------------------------------------------

typedef __bf16 bf16;
typedef __attribute__((ext_vector_type(16))) __bf16 v16bf;
typedef __attribute__((ext_vector_type(8)))  float  v8f;

constexpr int Bk   = 8;
constexpr int Nn   = 1024;
constexpr int Fin  = 32;
constexpr int Tt   = 16;
constexpr int Fc   = 64;      // F_CHEB
constexpr int Ftm  = 64;      // F_TIME
constexpr int TOUT = 8;       // T / STRIDE
constexpr int Ee   = 32768;
constexpr int ENN  = Ee + Nn; // edges + self loops
constexpr float LNEPS = 1e-5f;

// ------------------------- workspace layout (bytes) ------------------------
constexpr size_t SZ_NN_F32   = (size_t)Bk * Nn * Nn * 4;        // 33,554,432
constexpr size_t SZ_NN_BF16  = (size_t)Bk * Nn * Nn * 2;        // 16,777,216
constexpr size_t SZ_BTNF_F32 = (size_t)Bk * Tt * Nn * Fin * 4;  // 16,777,216
constexpr size_t SZ_BTNF_BF  = (size_t)Bk * Tt * Nn * Fin * 2;  //  8,388,608

constexpr size_t OFF_S     = 0;                       // S f32 -> spA f32 -> (reused) sg f32
constexpr size_t OFF_SIG   = OFF_S    + SZ_NN_F32;    // sigS bf16 -> spA bf16
constexpr size_t OFF_XTAT  = OFF_SIG  + SZ_NN_BF16;   // x_TAt f32 -> (reused) Tx2 f32
constexpr size_t OFF_TX0   = OFF_XTAT + SZ_BTNF_F32;  // Tx0 f32
constexpr size_t OFF_TX1   = OFF_TX0  + SZ_BTNF_F32;  // Tx1 f32
constexpr size_t OFF_XTBF  = OFF_TX1  + SZ_BTNF_F32;  // xT bf16 (B,T,N,F)
constexpr size_t OFF_XTF   = OFF_XTBF + SZ_BTNF_BF;   // xT f32  (B,T,N,F)
constexpr size_t OFF_RHS_T = OFF_XTF  + SZ_BTNF_F32;  // (B,N,T) f32
constexpr size_t OFF_LHS_T = OFF_RHS_T + (size_t)Bk*Nn*Tt*4;  // (B,T,N)
constexpr size_t OFF_TMPFT = OFF_LHS_T + (size_t)Bk*Tt*Nn*4;  // (B,F,T)
constexpr size_t OFF_ENORM = OFF_TMPFT + (size_t)Bk*Fin*Tt*4; // (B,T,T)
constexpr size_t OFF_LHS_S = OFF_ENORM + (size_t)Bk*Tt*Tt*4;  // (B,N,T)
constexpr size_t OFF_RHS_S = OFF_LHS_S + (size_t)Bk*Nn*Tt*4;  // (B,T,N)
constexpr size_t OFF_VSBF  = OFF_RHS_S + (size_t)Bk*Tt*Nn*4;  // Vs bf16 (N,N)
constexpr size_t OFF_DEG   = OFF_VSBF  + (size_t)Nn*Nn*2;     // (N) f32
constexpr size_t OFF_FW    = OFF_DEG   + (size_t)Nn*4;        // (E+N) f32
constexpr size_t OFF_FROW  = OFF_FW    + (size_t)ENN*4;       // (E+N) i32
constexpr size_t OFF_FCOL  = OFF_FROW  + (size_t)ENN*4;       // (E+N) i32
constexpr size_t OFF_ATT   = OFF_FCOL  + (size_t)ENN*4;       // (B,E+N) f32

// ------------------------------ helpers ------------------------------------
__device__ __forceinline__ bf16 f2bf(float f) {
  union { float f; unsigned u; } a; a.f = f;
  unsigned r = a.u + 0x7FFFu + ((a.u >> 16) & 1u);    // round to nearest even
  union { unsigned short s; bf16 b; } c;
  c.s = (unsigned short)(r >> 16);
  return c.b;
}

// A fragment (16x32 bf16, row-major source):
// lane L: row = base_row + (L&15); hi = L>>4; bf16 pairs at K offsets
// {0,2,4,6}+8*hi and {16,18,20,22}+8*hi  => two 16B loads.
__device__ __forceinline__ v16bf load_frag_a(const bf16* rowp, int hi) {
  union { uint4 u[2]; v16bf v; } f;
  f.u[0] = *reinterpret_cast<const uint4*>(rowp + 8 * hi);
  f.u[1] = *reinterpret_cast<const uint4*>(rowp + 16 + 8 * hi);
  return f.v;
}
// B fragment (32x16 bf16): lane = column (L&15); K rows {0..7}+8*hi and {16..23}+8*hi.
__device__ __forceinline__ v16bf load_frag_b(const bf16* p, int ld, int hi) {
  v16bf f;
  #pragma unroll
  for (int s = 0; s < 8; ++s) {
    f[s] = p[(size_t)(8 * hi + s) * ld];
    f[s + 8] = p[(size_t)(16 + 8 * hi + s) * ld];
  }
  return f;
}

#define WMMA_BF16(A, Bf, C) \
  __builtin_amdgcn_wmma_f32_16x16x32_bf16(false, (A), false, (Bf), (short)0, (C), false, false)

// ---------------------------------------------------------------------------
// K0: generic zero fill
__global__ void k_zero_f32(float* p, int n) {
  int i = blockIdx.x * blockDim.x + threadIdx.x;
  if (i < n) p[i] = 0.0f;
}

// K1: rhs_t[b,n,t] = sum_f U3[f] * x[b,n,f,t]
__global__ void k_rhs_t(const float* __restrict__ x, const float* __restrict__ U3,
                        float* __restrict__ rhs_t) {
  int idx = blockIdx.x * blockDim.x + threadIdx.x;
  if (idx >= Bk * Nn * Tt) return;
  int t = idx % Tt; int bn = idx / Tt;
  const float* xp = x + (size_t)bn * Fin * Tt + t;
  float s = 0.f;
  #pragma unroll
  for (int f = 0; f < Fin; ++f) s += U3[f] * xp[f * Tt];
  rhs_t[idx] = s;
}

// K2: tmpFT[b,f,t] = sum_n x[b,n,f,t] * U1[n]
__global__ void k_tmpft(const float* __restrict__ x, const float* __restrict__ U1,
                        float* __restrict__ tmpft) {
  int idx = blockIdx.x * blockDim.x + threadIdx.x;
  if (idx >= Bk * Fin * Tt) return;
  int t = idx % Tt; int f = (idx / Tt) % Fin; int b = idx / (Fin * Tt);
  const float* xp = x + (size_t)b * Nn * Fin * Tt + (size_t)f * Tt + t;
  float s = 0.f;
  for (int n = 0; n < Nn; ++n) s += xp[(size_t)n * Fin * Tt] * U1[n];
  tmpft[idx] = s;
}

// K3: lhs_t[b,t,m] = sum_f tmpFT[b,f,t] * U2[f,m]
__global__ void k_lhs_t(const float* __restrict__ tmpft, const float* __restrict__ U2,
                        float* __restrict__ lhs_t) {
  int idx = blockIdx.x * blockDim.x + threadIdx.x;
  if (idx >= Bk * Tt * Nn) return;
  int m = idx % Nn; int t = (idx / Nn) % Tt; int b = idx / (Tt * Nn);
  const float* tp = tmpft + (size_t)b * Fin * Tt + t;
  float s = 0.f;
  #pragma unroll
  for (int f = 0; f < Fin; ++f) s += tp[f * Tt] * U2[f * Nn + m];
  lhs_t[idx] = s;
}

// K4: prod_t -> sigmoid(+be) -> Ve mix -> softmax(axis=t) => E_norm (B,T,T)
__global__ __launch_bounds__(256) void k_temporal_attn(
    const float* __restrict__ lhs_t, const float* __restrict__ rhs_t,
    const float* __restrict__ be, const float* __restrict__ Ve,
    float* __restrict__ En) {
  int b = blockIdx.x;
  int t = threadIdx.x >> 4, u = threadIdx.x & 15;
  __shared__ float sig[Tt][Tt + 1];
  __shared__ float eat[Tt][Tt + 1];
  const float* L = lhs_t + (size_t)b * Tt * Nn;  // [t][n]
  const float* R = rhs_t + (size_t)b * Nn * Tt;  // [n][u]
  float p = 0.f;
  for (int n = 0; n < Nn; ++n) p += L[t * Nn + n] * R[n * Tt + u];
  sig[t][u] = 1.f / (1.f + expf(-(p + be[t * Tt + u])));
  __syncthreads();
  float e = 0.f;
  #pragma unroll
  for (int s = 0; s < Tt; ++s) e += Ve[t * Tt + s] * sig[s][u];
  eat[t][u] = e;
  __syncthreads();
  float mx = -1e30f;
  #pragma unroll
  for (int s = 0; s < Tt; ++s) mx = fmaxf(mx, eat[s][u]);
  float sum = 0.f;
  #pragma unroll
  for (int s = 0; s < Tt; ++s) sum += expf(eat[s][u] - mx);
  En[((size_t)b * Tt + t) * Tt + u] = expf(eat[t][u] - mx) / sum;
}

// K5: x_TAt[b,n,f,t] = sum_s x[b,n,f,s] * E_norm[b,s,t]
__global__ void k_xtat(const float* __restrict__ x, const float* __restrict__ En,
                       float* __restrict__ xtat) {
  int idx = blockIdx.x * blockDim.x + threadIdx.x;
  if (idx >= Bk * Nn * Fin * Tt) return;
  int t = idx % Tt; int bnf = idx / Tt; int b = idx / (Nn * Fin * Tt);
  const float* xp = x + (size_t)bnf * Tt;
  const float* ep = En + (size_t)b * Tt * Tt + t;
  float s = 0.f;
  #pragma unroll
  for (int q = 0; q < Tt; ++q) s += xp[q] * ep[q * Tt];
  xtat[idx] = s;
}

// K6: lhs_s[b,n,u] = sum_f (sum_t x_TAt[b,n,f,t]*W1[t]) * W2[f,u]
__global__ void k_lhs_s(const float* __restrict__ xtat, const float* __restrict__ W1,
                        const float* __restrict__ W2, float* __restrict__ lhs_s) {
  int idx = blockIdx.x * blockDim.x + threadIdx.x;
  if (idx >= Bk * Nn) return;
  const float* xp = xtat + (size_t)idx * Fin * Tt;
  float tmp[Fin];
  #pragma unroll
  for (int f = 0; f < Fin; ++f) {
    float s = 0.f;
    #pragma unroll
    for (int t = 0; t < Tt; ++t) s += xp[f * Tt + t] * W1[t];
    tmp[f] = s;
  }
  float* op = lhs_s + (size_t)idx * Tt;
  #pragma unroll
  for (int u = 0; u < Tt; ++u) {
    float s = 0.f;
    #pragma unroll
    for (int f = 0; f < Fin; ++f) s += tmp[f] * W2[f * Tt + u];
    op[u] = s;
  }
}

// K7: rhs_s[b,t,m] = sum_f W3[f] * x_TAt[b,m,f,t]
__global__ void k_rhs_s(const float* __restrict__ xtat, const float* __restrict__ W3,
                        float* __restrict__ rhs_s) {
  int idx = blockIdx.x * blockDim.x + threadIdx.x;
  if (idx >= Bk * Tt * Nn) return;
  int m = idx % Nn; int t = (idx / Nn) % Tt; int b = idx / (Tt * Nn);
  const float* xp = xtat + ((size_t)(b * Nn + m)) * Fin * Tt + t;
  float s = 0.f;
  #pragma unroll
  for (int f = 0; f < Fin; ++f) s += W3[f] * xp[f * Tt];
  rhs_s[idx] = s;
}

// K8: sigS[b,n,m] = sigmoid(sum_t lhs_s[b,n,t]*rhs_s[b,t,m] + bs[n,m])  (bf16 out)
__global__ void k_spatial_sig(const float* __restrict__ lhs_s,
                              const float* __restrict__ rhs_s,
                              const float* __restrict__ bs, bf16* __restrict__ sig) {
  int idx = blockIdx.x * blockDim.x + threadIdx.x;
  if (idx >= Bk * Nn * Nn) return;
  int rem = idx % (Nn * Nn);
  int b = idx / (Nn * Nn);
  int n = rem / Nn, m = rem % Nn;
  const float* L = lhs_s + ((size_t)b * Nn + n) * Tt;
  const float* R = rhs_s + (size_t)b * Tt * Nn + m;
  float p = 0.f;
  #pragma unroll
  for (int t = 0; t < Tt; ++t) p += L[t] * R[t * Nn];
  float s = 1.f / (1.f + expf(-(p + bs[rem])));
  sig[(size_t)idx] = f2bf(s);
}

// K8b: f32 -> bf16 convert
__global__ void k_f32_to_bf16(const float* __restrict__ src, bf16* __restrict__ dst, int n) {
  int i = blockIdx.x * blockDim.x + threadIdx.x;
  if (i < n) dst[i] = f2bf(src[i]);
}

// K9: xT[b,t,n,f] = x[b,n,f,t]  (bf16 and f32 copies)
__global__ void k_build_xt(const float* __restrict__ x, bf16* __restrict__ xtb,
                           float* __restrict__ xtf) {
  int idx = blockIdx.x * blockDim.x + threadIdx.x;
  if (idx >= Bk * Tt * Nn * Fin) return;
  int f = idx % Fin; int n = (idx / Fin) % Nn;
  int t = (idx / (Fin * Nn)) % Tt; int b = idx / (Fin * Nn * Tt);
  float v = x[(((size_t)(b * Nn + n)) * Fin + f) * Tt + t];
  xtb[idx] = f2bf(v);
  xtf[idx] = v;
}

// K10: S[b,i,m] = sum_n Vs[i,n] * sig[b,n,m]   (bf16 WMMA, f32 accum)
// 2x2 register blocking: each wave computes a 32x32 tile (4 WMMAs per K-step,
// 1 fragment load per WMMA). grid (N/32, N/(32*8), B), block 256 (8 waves).
__global__ __launch_bounds__(256) void k_gemm_vs_sig(
    const bf16* __restrict__ Vsb, const bf16* __restrict__ sig, float* __restrict__ S) {
  const int lane = threadIdx.x & 31;
  const int wave = threadIdx.x >> 5;
  const int b = blockIdx.z;
  const int i0 = blockIdx.x * 32;
  const int m0 = (blockIdx.y * 8 + wave) * 32;
  const int hi = lane >> 4, lrow = lane & 15;
  const bf16* Bm = sig + (size_t)b * Nn * Nn;
  const bf16* arow0 = Vsb + (size_t)(i0 + lrow) * Nn;
  const bf16* arow1 = arow0 + (size_t)16 * Nn;
  v8f a00 = {}, a01 = {}, a10 = {}, a11 = {};
  for (int k0 = 0; k0 < Nn; k0 += 32) {
    v16bf fa0 = load_frag_a(arow0 + k0, hi);
    v16bf fa1 = load_frag_a(arow1 + k0, hi);
    const bf16* brow = Bm + (size_t)k0 * Nn + m0 + lrow;
    v16bf fb0 = load_frag_b(brow, Nn, hi);
    v16bf fb1 = load_frag_b(brow + 16, Nn, hi);
    a00 = WMMA_BF16(fa0, fb0, a00);
    a01 = WMMA_BF16(fa0, fb1, a01);
    a10 = WMMA_BF16(fa1, fb0, a10);
    a11 = WMMA_BF16(fa1, fb1, a11);
  }
  float* Srow = S + (size_t)b * Nn * Nn;
  int r0 = i0 + hi * 8, c0 = m0 + lrow;
  #pragma unroll
  for (int j = 0; j < 8; ++j) {
    Srow[(size_t)(r0 + j) * Nn + c0]           = a00[j];
    Srow[(size_t)(r0 + j) * Nn + c0 + 16]      = a01[j];
    Srow[(size_t)(r0 + 16 + j) * Nn + c0]      = a10[j];
    Srow[(size_t)(r0 + 16 + j) * Nn + c0 + 16] = a11[j];
  }
}

// K11: spA = softmax over axis=1 (rows i) per (b, m) column; f32 in place + bf16 copy
__global__ void k_softmax_col(float* __restrict__ S, bf16* __restrict__ spAb) {
  int idx = blockIdx.x * blockDim.x + threadIdx.x;
  if (idx >= Bk * Nn) return;
  int b = idx / Nn, m = idx % Nn;
  float* col = S + (size_t)b * Nn * Nn + m;
  float mx = -1e30f;
  for (int i = 0; i < Nn; ++i) mx = fmaxf(mx, col[(size_t)i * Nn]);
  float sum = 0.f;
  for (int i = 0; i < Nn; ++i) sum += expf(col[(size_t)i * Nn] - mx);
  float inv = 1.f / sum;
  bf16* bcol = spAb + (size_t)b * Nn * Nn + m;
  for (int i = 0; i < Nn; ++i) {
    float v = expf(col[(size_t)i * Nn] - mx) * inv;
    col[(size_t)i * Nn] = v;
    bcol[(size_t)i * Nn] = f2bf(v);
  }
}

// K12b: degree count over original edges
__global__ void k_degree(const int* __restrict__ row, float* __restrict__ deg) {
  int e = blockIdx.x * blockDim.x + threadIdx.x;
  if (e < Ee) atomicAdd(&deg[row[e]], 1.0f);
}

// K12c: scaled-Laplacian weights; scale = 2/lambda_max = 1; diag weight = scale-1 = 0
__global__ void k_edge_weights(const int* __restrict__ row, const int* __restrict__ col,
                               const float* __restrict__ deg, float* __restrict__ fw,
                               int* __restrict__ frow, int* __restrict__ fcol) {
  int e = blockIdx.x * blockDim.x + threadIdx.x;
  if (e >= ENN) return;
  const float scale = 1.0f;  // 2 / LAMBDA_MAX
  if (e < Ee) {
    int r = row[e], c = col[e];
    float dr = deg[r], dc = deg[c];
    float ir = dr > 0.f ? rsqrtf(dr) : 0.f;
    float ic = dc > 0.f ? rsqrtf(dc) : 0.f;
    fw[e] = -scale * ir * ic;
    frow[e] = r; fcol[e] = c;
  } else {
    int d = e - Ee;
    fw[e] = scale - 1.0f;
    frow[e] = d; fcol[e] = d;
  }
}

// K13: att_norm[b,e] = fw[e] * spA[b, frow[e], fcol[e]]
__global__ void k_att_norm(const float* __restrict__ fw, const int* __restrict__ frow,
                           const int* __restrict__ fcol, const float* __restrict__ spA,
                           float* __restrict__ att) {
  int idx = blockIdx.x * blockDim.x + threadIdx.x;
  if (idx >= Bk * ENN) return;
  int e = idx % ENN, b = idx / ENN;
  att[idx] = fw[e] * spA[(size_t)b * Nn * Nn + (size_t)frow[e] * Nn + fcol[e]];
}

// K14: Tx0[b,t,i,f] = sum_j spA[b,i,j] * xT[b,t,j,f]   (bf16 WMMA)
// 2x2 blocking: wave computes 32 rows x all 32 f-cols; A frags reused across
// both f-tiles. grid (N/(32*8), T, B), block 256 (8 waves).
__global__ __launch_bounds__(256) void k_gemm_spa_x(
    const bf16* __restrict__ spAb, const bf16* __restrict__ xtb, float* __restrict__ Tx0) {
  const int lane = threadIdx.x & 31;
  const int wave = threadIdx.x >> 5;
  const int b = blockIdx.z, t = blockIdx.y;
  const int i0 = (blockIdx.x * 8 + wave) * 32;
  const int hi = lane >> 4, lrow = lane & 15;
  const bf16* A = spAb + (size_t)b * Nn * Nn;
  const bf16* Bm = xtb + ((size_t)(b * Tt + t)) * Nn * Fin;
  const bf16* arow0 = A + (size_t)(i0 + lrow) * Nn;
  const bf16* arow1 = arow0 + (size_t)16 * Nn;
  v8f a00 = {}, a01 = {}, a10 = {}, a11 = {};
  for (int k0 = 0; k0 < Nn; k0 += 32) {
    v16bf fa0 = load_frag_a(arow0 + k0, hi);
    v16bf fa1 = load_frag_a(arow1 + k0, hi);
    const bf16* brow = Bm + (size_t)k0 * Fin + lrow;
    v16bf fb0 = load_frag_b(brow, Fin, hi);
    v16bf fb1 = load_frag_b(brow + 16, Fin, hi);
    a00 = WMMA_BF16(fa0, fb0, a00);
    a01 = WMMA_BF16(fa0, fb1, a01);
    a10 = WMMA_BF16(fa1, fb0, a10);
    a11 = WMMA_BF16(fa1, fb1, a11);
  }
  float* orow = Tx0 + ((size_t)(b * Tt + t)) * Nn * Fin;
  int r0 = i0 + hi * 8, c0 = lrow;
  #pragma unroll
  for (int j = 0; j < 8; ++j) {
    orow[(size_t)(r0 + j) * Fin + c0]           = a00[j];
    orow[(size_t)(r0 + j) * Fin + c0 + 16]      = a01[j];
    orow[(size_t)(r0 + 16 + j) * Fin + c0]      = a10[j];
    orow[(size_t)(r0 + 16 + j) * Fin + c0 + 16] = a11[j];
  }
}

// K16/K18: scatter propagation  zout[b,t,fcol[e],:] += att[b,e] * zin[b,t,frow[e],:]
__global__ void k_prop_scatter(const float* __restrict__ att, const int* __restrict__ frow,
                               const int* __restrict__ fcol, const float* __restrict__ zin,
                               float* __restrict__ zout) {
  long long idx = (long long)blockIdx.x * blockDim.x + threadIdx.x;
  if (idx >= (long long)Bk * Tt * ENN) return;
  int e = (int)(idx % ENN);
  long long bt = idx / ENN;
  int b = (int)(bt / Tt);
  float a = att[(size_t)b * ENN + e];
  if (a == 0.0f) return;  // self-loop weight is (2/lmax - 1) == 0
  const float* src = zin + ((size_t)bt * Nn + frow[e]) * Fin;
  float* dst = zout + ((size_t)bt * Nn + fcol[e]) * Fin;
  #pragma unroll
  for (int f = 0; f < Fin; ++f) atomicAdd(dst + f, a * src[f]);
}

// K19: Tx2 = 2*P - Tx0   (P accumulated in Tx2's buffer)
__global__ void k_tx2_final(float* __restrict__ Tx2, const float* __restrict__ Tx0, int n) {
  int i = blockIdx.x * blockDim.x + threadIdx.x;
  if (i < n) Tx2[i] = 2.0f * Tx2[i] - Tx0[i];
}

// K20: sg[b,t,n,o] = relu(sum_k Tx_k[b,t,n,:] @ cheb_W[k] + cheb_b)
__global__ __launch_bounds__(256) void k_cheb_combine(
    const float* __restrict__ Tx0, const float* __restrict__ Tx1,
    const float* __restrict__ Tx2, const float* __restrict__ cW,
    const float* __restrict__ cb, float* __restrict__ sg) {
  __shared__ float sW[3 * Fin * Fc];
  __shared__ float sb[Fc];
  for (int i = threadIdx.x; i < 3 * Fin * Fc; i += 256) sW[i] = cW[i];
  for (int i = threadIdx.x; i < Fc; i += 256) sb[i] = cb[i];
  __syncthreads();
  int idx = blockIdx.x * 256 + threadIdx.x;  // (b*T + t)*N + n
  if (idx >= Bk * Tt * Nn) return;
  float x0[Fin], x1[Fin], x2[Fin];
  const float* r0 = Tx0 + (size_t)idx * Fin;
  const float* r1 = Tx1 + (size_t)idx * Fin;
  const float* r2 = Tx2 + (size_t)idx * Fin;
  #pragma unroll
  for (int f = 0; f < Fin; ++f) { x0[f] = r0[f]; x1[f] = r1[f]; x2[f] = r2[f]; }
  float* o = sg + (size_t)idx * Fc;
  for (int oc = 0; oc < Fc; ++oc) {
    float s = sb[oc];
    #pragma unroll
    for (int f = 0; f < Fin; ++f) {
      s += x0[f] * sW[f * Fc + oc]
         + x1[f] * sW[(Fin + f) * Fc + oc]
         + x2[f] * sW[(2 * Fin + f) * Fc + oc];
    }
    o[oc] = fmaxf(s, 0.0f);
  }
}

// K21: time conv (1,3)/stride2/pad1 + 1x1 residual conv + relu + LayerNorm(Ft) + transpose
// grid (N, TOUT, B), block = 64 (one thread per output channel)
__global__ __launch_bounds__(64) void k_tconv_ln(
    const float* __restrict__ sg, const float* __restrict__ x,
    const float* __restrict__ Wt, const float* __restrict__ bt,
    const float* __restrict__ Wr, const float* __restrict__ br,
    const float* __restrict__ lng, const float* __restrict__ lnb,
    float* __restrict__ out) {
  int n = blockIdx.x, tp = blockIdx.y, b = blockIdx.z;
  int o = threadIdx.x;
  float acc = bt[o] + br[o];
  #pragma unroll
  for (int kk = 0; kk < 3; ++kk) {
    int ts = 2 * tp + kk - 1;
    if (ts < 0 || ts >= Tt) continue;
    const float* sp = sg + (((size_t)(b * Tt + ts)) * Nn + n) * Fc;
    const float* wp = Wt + (size_t)o * Fc * 3 + kk;
    for (int c = 0; c < Fc; ++c) acc += sp[c] * wp[c * 3];
  }
  const float* xp = x + ((size_t)(b * Nn + n)) * Fin * Tt + 2 * tp;
  const float* wr = Wr + (size_t)o * Fin;
  #pragma unroll
  for (int c = 0; c < Fin; ++c) acc += xp[c * Tt] * wr[c];
  float v = fmaxf(acc, 0.0f);
  __shared__ float sv[Ftm];
  sv[o] = v;
  __syncthreads();
  float mu = 0.f;
  #pragma unroll
  for (int i = 0; i < Ftm; ++i) mu += sv[i];
  mu *= (1.0f / Ftm);
  float var = 0.f;
  #pragma unroll
  for (int i = 0; i < Ftm; ++i) { float d = sv[i] - mu; var += d * d; }
  var *= (1.0f / Ftm);
  float zn = (v - mu) * rsqrtf(var + LNEPS) * lng[o] + lnb[o];
  out[(((size_t)(b * Nn + n)) * Ftm + o) * TOUT + tp] = zn;
}

// ---------------------------------------------------------------------------
static inline int cdiv(long long a, int b) { return (int)((a + b - 1) / b); }

extern "C" void kernel_launch(void* const* d_in, const int* in_sizes, int n_in,
                              void* d_out, int out_size, void* d_ws, size_t ws_size,
                              hipStream_t stream) {
  (void)in_sizes; (void)n_in; (void)out_size; (void)ws_size;
  const float* x   = (const float*)d_in[0];
  const int*   ei  = (const int*)d_in[1];
  const float* U1  = (const float*)d_in[2];
  const float* U2  = (const float*)d_in[3];
  const float* U3  = (const float*)d_in[4];
  const float* be  = (const float*)d_in[5];
  const float* Ve  = (const float*)d_in[6];
  const float* W1  = (const float*)d_in[7];
  const float* W2  = (const float*)d_in[8];
  const float* W3  = (const float*)d_in[9];
  const float* bs  = (const float*)d_in[10];
  const float* Vs  = (const float*)d_in[11];
  const float* cW  = (const float*)d_in[12];
  const float* cb  = (const float*)d_in[13];
  const float* Wt  = (const float*)d_in[14];
  const float* bt  = (const float*)d_in[15];
  const float* Wr  = (const float*)d_in[16];
  const float* br  = (const float*)d_in[17];
  const float* lng = (const float*)d_in[18];
  const float* lnb = (const float*)d_in[19];
  float* out = (float*)d_out;

  char* ws = (char*)d_ws;
  float* S      = (float*)(ws + OFF_S);      // S -> spA (f32); later reused as sg
  float* sg     = (float*)(ws + OFF_S);
  bf16*  sigS   = (bf16*)(ws + OFF_SIG);     // sigS -> spA bf16
  bf16*  spAb   = (bf16*)(ws + OFF_SIG);
  float* xtat   = (float*)(ws + OFF_XTAT);   // x_TAt; later reused as Tx2
  float* Tx2    = (float*)(ws + OFF_XTAT);
  float* Tx0    = (float*)(ws + OFF_TX0);
  float* Tx1    = (float*)(ws + OFF_TX1);
  bf16*  xtb    = (bf16*)(ws + OFF_XTBF);
  float* xtf    = (float*)(ws + OFF_XTF);
  float* rhs_t  = (float*)(ws + OFF_RHS_T);
  float* lhs_t  = (float*)(ws + OFF_LHS_T);
  float* tmpft  = (float*)(ws + OFF_TMPFT);
  float* En     = (float*)(ws + OFF_ENORM);
  float* lhs_s  = (float*)(ws + OFF_LHS_S);
  float* rhs_s  = (float*)(ws + OFF_RHS_S);
  bf16*  Vsb    = (bf16*)(ws + OFF_VSBF);
  float* deg    = (float*)(ws + OFF_DEG);
  float* fw     = (float*)(ws + OFF_FW);
  int*   frow   = (int*)(ws + OFF_FROW);
  int*   fcol   = (int*)(ws + OFF_FCOL);
  float* att    = (float*)(ws + OFF_ATT);

  const int* erow = ei;
  const int* ecol = ei + Ee;

  // ---- temporal attention ----
  k_rhs_t<<<cdiv((long long)Bk * Nn * Tt, 256), 256, 0, stream>>>(x, U3, rhs_t);
  k_tmpft<<<cdiv((long long)Bk * Fin * Tt, 256), 256, 0, stream>>>(x, U1, tmpft);
  k_lhs_t<<<cdiv((long long)Bk * Tt * Nn, 256), 256, 0, stream>>>(tmpft, U2, lhs_t);
  k_temporal_attn<<<Bk, 256, 0, stream>>>(lhs_t, rhs_t, be, Ve, En);
  k_xtat<<<cdiv((long long)Bk * Nn * Fin * Tt, 256), 256, 0, stream>>>(x, En, xtat);

  // ---- spatial attention ----
  k_lhs_s<<<cdiv((long long)Bk * Nn, 256), 256, 0, stream>>>(xtat, W1, W2, lhs_s);
  k_rhs_s<<<cdiv((long long)Bk * Tt * Nn, 256), 256, 0, stream>>>(xtat, W3, rhs_s);
  k_spatial_sig<<<cdiv((long long)Bk * Nn * Nn, 256), 256, 0, stream>>>(lhs_s, rhs_s, bs, sigS);
  k_f32_to_bf16<<<cdiv((long long)Nn * Nn, 256), 256, 0, stream>>>(Vs, Vsb, Nn * Nn);

  // big GEMM 1: S = Vs @ sigS  (WMMA bf16, 2x2 blocked)
  {
    dim3 grid(Nn / 32, Nn / (32 * 8), Bk);
    k_gemm_vs_sig<<<grid, 256, 0, stream>>>(Vsb, sigS, S);
  }
  // softmax over rows (axis=1) -> spA (f32 in place, bf16 copy overwrites sigS)
  k_softmax_col<<<cdiv((long long)Bk * Nn, 256), 256, 0, stream>>>(S, spAb);

  // ---- scaled-Laplacian edge weights + attention gather ----
  k_zero_f32<<<cdiv(Nn, 256), 256, 0, stream>>>(deg, Nn);
  k_degree<<<cdiv(Ee, 256), 256, 0, stream>>>(erow, deg);
  k_edge_weights<<<cdiv(ENN, 256), 256, 0, stream>>>(erow, ecol, deg, fw, frow, fcol);
  k_att_norm<<<cdiv((long long)Bk * ENN, 256), 256, 0, stream>>>(fw, frow, fcol, S, att);

  // ---- Chebyshev recursion ----
  k_build_xt<<<cdiv((long long)Bk * Tt * Nn * Fin, 256), 256, 0, stream>>>(x, xtb, xtf);
  {
    dim3 grid(Nn / (32 * 8), Tt, Bk);   // big GEMM 2: Tx0 = spA @ xT (WMMA bf16, 2x2 blocked)
    k_gemm_spa_x<<<grid, 256, 0, stream>>>(spAb, xtb, Tx0);
  }
  const int ntxf = Bk * Tt * Nn * Fin;
  k_zero_f32<<<cdiv(ntxf, 256), 256, 0, stream>>>(Tx1, ntxf);
  k_prop_scatter<<<cdiv((long long)Bk * Tt * ENN, 256), 256, 0, stream>>>(att, frow, fcol, xtf, Tx1);
  k_zero_f32<<<cdiv(ntxf, 256), 256, 0, stream>>>(Tx2, ntxf);
  k_prop_scatter<<<cdiv((long long)Bk * Tt * ENN, 256), 256, 0, stream>>>(att, frow, fcol, Tx1, Tx2);
  k_tx2_final<<<cdiv(ntxf, 256), 256, 0, stream>>>(Tx2, Tx0, ntxf);
  k_cheb_combine<<<cdiv((long long)Bk * Tt * Nn, 256), 256, 0, stream>>>(Tx0, Tx1, Tx2, cW, cb, sg);

  // ---- time conv + residual + relu + LayerNorm + final transpose ----
  {
    dim3 grid(Nn, TOUT, Bk);
    k_tconv_ln<<<grid, Ftm, 0, stream>>>(sg, x, Wt, bt, Wr, br, lng, lnb, out);
  }
}
